// CoreRelu_83863531422003
// MI455X (gfx1250) — matrix operand order repaired and checked by reading, not compiled, not run on hardware
//
#include <hip/hip_runtime.h>

typedef __attribute__((ext_vector_type(16))) __bf16 bf16x16;
typedef __attribute__((ext_vector_type(8)))  __bf16 bf16x8;
typedef __attribute__((ext_vector_type(8)))  float  f32x8;

#define LDA 40   // halves per A-row in LDS (padded; conflict-free frag loads)

__device__ __forceinline__ unsigned short f32_to_bf16(float f) {
    unsigned int u = __float_as_uint(f);
    u += 0x7FFFu + ((u >> 16) & 1u);          // round-to-nearest-even
    return (unsigned short)(u >> 16);
}
__device__ __forceinline__ float bf16_to_f32(unsigned short h) {
    return __uint_as_float(((unsigned int)h) << 16);
}

// ---------------- weight transpose + fp32->bf16 convert:  Wt[n][k] = bf16(W[k][n]) ----
__global__ void transpose_bf16(const float* __restrict__ W, unsigned short* __restrict__ Wt,
                               int K, int N) {
    __shared__ float tile[32][33];
    int n0 = blockIdx.x * 32, k0 = blockIdx.y * 32;
    int tx = threadIdx.x, ty = threadIdx.y;
    #pragma unroll
    for (int i = 0; i < 32; i += 8)
        tile[ty + i][tx] = W[(size_t)(k0 + ty + i) * N + n0 + tx];
    __syncthreads();
    #pragma unroll
    for (int i = 0; i < 32; i += 8)
        Wt[(size_t)(n0 + ty + i) * K + k0 + tx] = f32_to_bf16(tile[tx][ty + i]);
}

// ---------------- fused GEMM (+bias+relu  or  +LayerNorm+relu), bf16 out --------------
// Block: 256 thr (8 wave32). Computes 32 rows x 1024 cols.  K in steps of 32.
// B tiles double-buffered in LDS, filled by GLOBAL_LOAD_ASYNC_TO_LDS_B128 (ASYNCcnt),
// stored in WMMA fragment order: [tile(64)][lg(2)][lane(16)][16 K-halves] = 64KB.
template<bool LN, bool AF32>
__global__ __launch_bounds__(256, 1)
void mlp_layer(const float* __restrict__ Xf,            // fp32 activations (AF32 only)
               unsigned short* __restrict__ H,          // bf16 activations [N][1024] (in/out)
               const unsigned short* __restrict__ Wt,   // bf16 weights, transposed [1024][K]
               const float* __restrict__ beta_or_bias,  // LN beta, or layer-0 bias
               const float* __restrict__ gamma,         // LN gamma (LN only)
               int K)
{
    __shared__ __align__(16) unsigned short ldsB[2 * 32768]; // 2 x 64KB (buf0 reused as staging)
    __shared__ __align__(16) unsigned short ldsA[2 * 32 * LDA];
    __shared__ float red[32][2];                             // row sum / sumsq

    const int t    = threadIdx.x;
    const int lane = t & 31;
    const int wv   = t >> 5;        // wave 0..7
    const int lg   = lane >> 4;     // half-wave group 0/1
    const int ln   = lane & 15;
    const int rowBase = blockIdx.x * 32;
    const unsigned ldsB_base = (unsigned)(uintptr_t)&ldsB[0];  // raw LDS byte offset

    if (LN && t < 64) ((float*)red)[t] = 0.f;

    f32x8 acc[2][8];
    #pragma unroll
    for (int a = 0; a < 2; ++a)
        #pragma unroll
        for (int b = 0; b < 8; ++b)
            acc[a][b] = (f32x8){0.f,0.f,0.f,0.f,0.f,0.f,0.f,0.f};

    const int nsteps = K >> 5;

    // --- async issue of one 1024x32 B tile into LDS buffer `buf` (fragment order) ---
    auto issueB = [&](int ks, int buf) {
        const int kbase = ks << 5;
        const unsigned base = ldsB_base + (unsigned)buf * 65536u;
        #pragma unroll
        for (int i = 0; i < 16; ++i) {
            int c   = t + (i << 8);          // 0..4095 chunks of 16B
            int n   = c >> 2;                // weight column 0..1023
            int lgi = (c >> 1) & 1;          // K half-group
            int hf  = c & 1;                 // which 16B of the 32B run
            const unsigned short* src = Wt + (size_t)n * K + kbase + lgi * 8 + hf * 16;
            unsigned loff = base + (unsigned)((((n >> 4) * 2 + lgi) * 16 + (n & 15)) * 32 + hf * 16);
            asm volatile("global_load_async_to_lds_b128 %0, %1, off"
                         :: "v"(loff), "v"((unsigned long long)(uintptr_t)src)
                         : "memory");
        }
    };
    // --- normal (converting) load of the 32x32 A tile into LDS buffer `buf` ---
    auto loadA = [&](int ks, int buf) {
        const int kbase = ks << 5;
        const int abase = buf * 32 * LDA;
        if (AF32) {
            int r = t >> 3, kk = (t & 7) << 2;
            const float* src = Xf + (size_t)(rowBase + r) * K + kbase + kk;
            const float4 v = *reinterpret_cast<const float4*>(src);
            ushort4 h;
            h.x = f32_to_bf16(v.x); h.y = f32_to_bf16(v.y);
            h.z = f32_to_bf16(v.z); h.w = f32_to_bf16(v.w);
            *reinterpret_cast<ushort4*>(&ldsA[abase + r * LDA + kk]) = h;
            __builtin_prefetch(src + 32, 0, 0);        // global_prefetch_b8
        } else if (t < 128) {
            int r = t >> 2, kk = (t & 3) << 3;
            const unsigned short* src = H + (size_t)(rowBase + r) * 1024 + kbase + kk;
            uint4 v = *reinterpret_cast<const uint4*>(src);
            *reinterpret_cast<uint4*>(&ldsA[abase + r * LDA + kk]) = v;
            __builtin_prefetch(src + 32, 0, 0);
        }
    };

    issueB(0, 0);
    loadA(0, 0);

    for (int ks = 0; ks < nsteps; ++ks) {
        const int buf  = ks & 1;
        const bool more = (ks + 1 < nsteps);
        if (more) {                       // prefetch next tile while computing this one
            issueB(ks + 1, buf ^ 1);
            loadA(ks + 1, buf ^ 1);
            asm volatile("s_wait_asynccnt 0x10" ::: "memory");  // current buf's 16 retired
        } else {
            asm volatile("s_wait_asynccnt 0x0" ::: "memory");
        }
        __syncthreads();                  // publish buf to all waves

        // A frag (16x32 bf16): lanes 0-15 hold K{0-7,16-23}, lanes 16-31 K{8-15,24-31}
        const int abase = buf * 32 * LDA;
        bf16x16 afrag[2];
        #pragma unroll
        for (int tm = 0; tm < 2; ++tm) {
            int m = tm * 16 + ln;
            bf16x8 lo = *reinterpret_cast<const bf16x8*>(&ldsA[abase + m * LDA + lg * 8]);
            bf16x8 hi = *reinterpret_cast<const bf16x8*>(&ldsA[abase + m * LDA + lg * 8 + 16]);
            afrag[tm] = __builtin_shufflevector(lo, hi,
                0,1,2,3,4,5,6,7,8,9,10,11,12,13,14,15);
        }
        // B frag: [tile][lg][ln][16 halves] -> one contiguous 32B LDS load per lane.
        // 3-slot rotating pipeline, prefetch distance 2: slot (tn+2)%3 is filled while
        // slot tn%3 feeds the WMMAs; no register WAR, so ds_loads stay 2 frags ahead.
        const int bbase  = buf * 32768;
        const int bfbase = bbase + (lg * 16 + ln) * 16;      // + tile*512 halves
        bf16x16 bf[3];
        bf[0] = *reinterpret_cast<const bf16x16*>(&ldsB[bfbase + (wv * 8 + 0) * 512]);
        bf[1] = *reinterpret_cast<const bf16x16*>(&ldsB[bfbase + (wv * 8 + 1) * 512]);
        #pragma unroll
        for (int tn = 0; tn < 8; ++tn) {
            if (tn + 2 < 8)
                bf[(tn + 2) % 3] = *reinterpret_cast<const bf16x16*>(
                    &ldsB[bfbase + (wv * 8 + tn + 2) * 512]);
            acc[0][tn] = __builtin_amdgcn_wmma_f32_16x16x32_bf16(
                false, afrag[0], false, bf[tn % 3], (short)0, acc[0][tn], false, false);
            acc[1][tn] = __builtin_amdgcn_wmma_f32_16x16x32_bf16(
                false, afrag[1], false, bf[tn % 3], (short)0, acc[1][tn], false, false);
        }
        __syncthreads();                  // all waves done reading buf -> refill allowed
    }

    // ---- LayerNorm row statistics (full 1024-wide rows live in this block) ----
    if (LN) {
        #pragma unroll
        for (int tm = 0; tm < 2; ++tm) {
            #pragma unroll
            for (int j = 0; j < 8; ++j) {
                float s = 0.f, q = 0.f;
                #pragma unroll
                for (int tn = 0; tn < 8; ++tn) {
                    float v = acc[tm][tn][j];
                    s += v; q += v * v;
                }
                #pragma unroll
                for (int off = 8; off >= 1; off >>= 1) {   // reduce 16 lanes sharing a row
                    s += __shfl_xor(s, off, 32);
                    q += __shfl_xor(q, off, 32);
                }
                if (ln == 0) {
                    int r = tm * 16 + lg * 8 + j;
                    atomicAdd(&red[r][0], s);   // ds_add_f32
                    atomicAdd(&red[r][1], q);
                }
            }
        }
        __syncthreads();
    }

    // ---- epilogue: (LN?) scale+shift, relu, bf16 -> LDS staging (aliases B buf0) ----
    #pragma unroll
    for (int tm = 0; tm < 2; ++tm) {
        #pragma unroll
        for (int j = 0; j < 8; ++j) {
            int r = tm * 16 + lg * 8 + j;
            float mu = 0.f, rs = 1.f;
            if (LN) {
                mu = red[r][0] * (1.f / 1024.f);
                float var = red[r][1] * (1.f / 1024.f) - mu * mu;
                rs = __frsqrt_rn(var + 1e-6f);
            }
            #pragma unroll
            for (int tn = 0; tn < 8; ++tn) {
                int col = (wv * 8 + tn) * 16 + ln;
                float v = acc[tm][tn][j];
                float o = LN ? ((v - mu) * rs * gamma[col] + beta_or_bias[col])
                             : (v + beta_or_bias[col]);
                ldsB[r * 1024 + col] = f32_to_bf16(fmaxf(o, 0.f));
            }
        }
    }
    __syncthreads();
    // ---- coalesced 16B stores of the 32x1024 bf16 tile ----
    #pragma unroll
    for (int i = 0; i < 16; ++i) {
        int c  = t + (i << 8);         // 0..4095
        int r  = c >> 7;
        int cc = (c & 127) << 3;
        uint4 v = *reinterpret_cast<const uint4*>(&ldsB[r * 1024 + cc]);
        *reinterpret_cast<uint4*>(H + (size_t)(rowBase + r) * 1024 + cc) = v;
    }
}

// ---------------- final Dense(1) + relu: one wave per row ----------------------------
__global__ __launch_bounds__(256)
void out_kernel(const unsigned short* __restrict__ H, const float* __restrict__ Wout,
                const float* __restrict__ bout, float* __restrict__ out)
{
    int row  = blockIdx.x * 8 + (threadIdx.x >> 5);
    int lane = threadIdx.x & 31;
    const unsigned short* hr = H + (size_t)row * 1024;
    float s = 0.f;
    #pragma unroll
    for (int i = 0; i < 4; ++i) {
        int c = (i * 32 + lane) * 8;
        uint4 pk = *reinterpret_cast<const uint4*>(hr + c);
        const unsigned short* hs = reinterpret_cast<const unsigned short*>(&pk);
        #pragma unroll
        for (int j = 0; j < 8; ++j)
            s += bf16_to_f32(hs[j]) * Wout[c + j];
    }
    #pragma unroll
    for (int off = 16; off >= 1; off >>= 1)
        s += __shfl_xor(s, off, 32);
    if (lane == 0) out[row] = fmaxf(s + bout[0], 0.f);
}

extern "C" void kernel_launch(void* const* d_in, const int* in_sizes, int n_in,
                              void* d_out, int out_size, void* d_ws, size_t ws_size,
                              hipStream_t stream) {
    const float* X    = (const float*)d_in[0];
    const float* W0   = (const float*)d_in[1];
    const float* b0   = (const float*)d_in[2];
    const float* W1   = (const float*)d_in[3];
    const float* g1   = (const float*)d_in[4];
    const float* be1  = (const float*)d_in[5];
    const float* W2   = (const float*)d_in[6];
    const float* g2   = (const float*)d_in[7];
    const float* be2  = (const float*)d_in[8];
    const float* W3   = (const float*)d_in[9];
    const float* g3   = (const float*)d_in[10];
    const float* be3  = (const float*)d_in[11];
    const float* Wout = (const float*)d_in[12];
    const float* bout = (const float*)d_in[13];
    float* out = (float*)d_out;

    char* ws = (char*)d_ws;
    const size_t MB = 1u << 20;
    unsigned short* Wt0 = (unsigned short*)(ws);            // 512*1024*2  = 1 MB
    unsigned short* Wt1 = (unsigned short*)(ws + 1 * MB);   // 1024*1024*2 = 2 MB
    unsigned short* Wt2 = (unsigned short*)(ws + 3 * MB);
    unsigned short* Wt3 = (unsigned short*)(ws + 5 * MB);
    unsigned short* H   = (unsigned short*)(ws + 7 * MB);   // 65536*1024*2 = 128 MB

    dim3 tb(32, 8);
    transpose_bf16<<<dim3(32, 16), tb, 0, stream>>>(W0, Wt0, 512, 1024);
    transpose_bf16<<<dim3(32, 32), tb, 0, stream>>>(W1, Wt1, 1024, 1024);
    transpose_bf16<<<dim3(32, 32), tb, 0, stream>>>(W2, Wt2, 1024, 1024);
    transpose_bf16<<<dim3(32, 32), tb, 0, stream>>>(W3, Wt3, 1024, 1024);

    mlp_layer<false, true ><<<2048, 256, 0, stream>>>(X, H, Wt0, b0, nullptr, 512);
    mlp_layer<true,  false><<<2048, 256, 0, stream>>>(nullptr, H, Wt1, be1, g1, 1024);
    mlp_layer<true,  false><<<2048, 256, 0, stream>>>(nullptr, H, Wt2, be2, g2, 1024);
    mlp_layer<true,  false><<<2048, 256, 0, stream>>>(nullptr, H, Wt3, be3, g3, 1024);

    out_kernel<<<65536 / 8, 256, 0, stream>>>(H, Wout, bout, out);
}
